// SparseLinearAttention_28174985462231
// MI455X (gfx1250) — compile-verified, matrix operand order checked by reading, not run on hardware
//
#include <hip/hip_runtime.h>
#include <math.h>

// ---------------------------------------------------------------------------
// SparseLinearAttention for MI455X (gfx1250, wave32, WMMA).
// B=1, T=2048, H=8, D=64, BLOCK=64 -> 32 q-blocks x 32 k-blocks per head.
// - All WMMA operands staged in LDS in *fragment order*: each operand is one
//   contiguous 32-byte per-lane read (2x ds_load_b128).
// - K/V/kv tiles are pre-converted to f16 fragment order ONCE per (h,kb) in
//   kv_kernel; the attention hot loop only copies 8KB tiles global->LDS,
//   using global_load_async_to_lds_b128 (ASYNCcnt) when available.
// ---------------------------------------------------------------------------

typedef __attribute__((ext_vector_type(16))) _Float16 v16h;
typedef __attribute__((ext_vector_type(8)))  _Float16 v8h;
typedef __attribute__((ext_vector_type(8)))  float    v8f;
typedef __attribute__((ext_vector_type(4)))  int      v4i;

#define Hh   8
#define Dd   64
#define NB   32      // nq == nk
#define BLK  64
#define EPSF 1e-6f
#define SCALEF 0.125f   // 1/sqrt(64), TEMP=1

// workspace layout (bytes)
#define WS_QREP 0                                   // NB*H*D f32
#define WS_KREP (WS_QREP + NB*Hh*Dd*4)              // NB*H*D f32
#define WS_KSUM (WS_KREP + NB*Hh*Dd*4)              // H*NB*D f32
#define WS_FLAG (WS_KSUM + Hh*NB*Dd*4)              // H*NB int
#define WS_KV   (WS_FLAG + Hh*NB*4)                 // H*NB*D*D f16 (kv, B-frag order)
#define WS_KF   (WS_KV + Hh*NB*Dd*Dd*2)             // H*NB*D*D f16 (K^T B-frag tiles)
#define WS_VF   (WS_KF + Hh*NB*Dd*Dd*2)             // H*NB*D*D f16 (V   B-frag tiles)

// ---------------------------------------------------------------------------
// Fragment-order mapping for f16 16x16x32 WMMA operands (ISA 7.12.2):
//  A tile 16(M)x32(K): lane = mloc + 16*hi,  hi = (kk>>3)&1,
//                      j = (kk&7) + 8*((kk>>4)&1)
//  B tile 32(K)x16(N): lane = n + 16*hi (symmetric mapping assumption).
// Storage: half offset = (frag*32 + lane)*16 + j.
// For an octet of 8 consecutive kk (kk0 = o*8):
//   kc = o>>2, hi = o&1, j0 = 8*((o>>1)&1)  -> 8 contiguous halves.
// ---------------------------------------------------------------------------
__device__ __forceinline__ int frag_chunk_off(int frag, int mn, int o) {
  return (frag * 32 + (mn + 16 * (o & 1))) * 16 + ((o >> 1) & 1) * 8;
}

__device__ __forceinline__ v8f wmma_f16(v16h a, v16h b, v8f c) {
  return __builtin_amdgcn_wmma_f32_16x16x32_f16(false, a, false, b, (short)0, c, false, false);
}

// ---- async global->LDS copy (CDNA5 ASYNCcnt path), with safe fallback -----
#if defined(__gfx1250__) && __has_builtin(__builtin_amdgcn_global_load_async_to_lds_b128)
#define HAVE_ASYNC_LDS 1
typedef __attribute__((address_space(1))) v4i gv4i_t;
typedef __attribute__((address_space(3))) v4i lv4i_t;
#endif

__device__ __forceinline__ void cp16_to_lds(void* lds_dst, const void* gsrc) {
#ifdef HAVE_ASYNC_LDS
  __builtin_amdgcn_global_load_async_to_lds_b128(
      (gv4i_t*)gsrc, (lv4i_t*)lds_dst, 0, 0);
#else
  *(uint4*)lds_dst = *(const uint4*)gsrc;
#endif
}

__device__ __forceinline__ void wait_async0() {
#ifdef HAVE_ASYNC_LDS
#if __has_builtin(__builtin_amdgcn_s_wait_asynccnt)
  __builtin_amdgcn_s_wait_asynccnt(0);
#else
  asm volatile("s_wait_asynccnt 0" ::: "memory");
#endif
#endif
}

// reductions across the 16 lanes of one half-wave (C-layout keeps row M in
// lanes 0-15 and row M+8 in lanes 16-31; masks 1..8 never cross the halves)
__device__ __forceinline__ float half_reduce_max(float x) {
#pragma unroll
  for (int m = 1; m <= 8; m <<= 1) x = fmaxf(x, __shfl_xor(x, m, 32));
  return x;
}
__device__ __forceinline__ float half_reduce_sum(float x) {
#pragma unroll
  for (int m = 1; m <= 8; m <<= 1) x += __shfl_xor(x, m, 32);
  return x;
}

// ---------------------------------------------------------------------------
// Kernel 1: block representatives (means).  grid 512 x 64 threads.
// ---------------------------------------------------------------------------
__global__ void rep_kernel(const float* __restrict__ q, const float* __restrict__ k,
                           float* __restrict__ ws) {
  const int bid = blockIdx.x;
  const int which = bid >> 8;     // 0 = q_rep, 1 = k_rep
  const int idx = bid & 255;      // qb*8 + h
  const int d = threadIdx.x;
  const float* src = which ? k : q;
  float* dst = (float*)((char*)ws + (which ? WS_KREP : WS_QREP));
  const int qb = idx >> 3, h = idx & 7;
  float s = 0.f;
  for (int t = 0; t < BLK; ++t)
    s += src[((qb * BLK + t) * Hh + h) * Dd + d];
  dst[idx * Dd + d] = s * (1.0f / BLK);
}

// ---------------------------------------------------------------------------
// Kernel 2: classification. One 256-thread workgroup; thread = (h,qb).
// ---------------------------------------------------------------------------
__global__ void classify_kernel(float* ws) {
  const int tid = threadIdx.x;          // 0..255
  const int h = tid >> 5, qb = tid & 31;
  const float* qr = (const float*)((char*)ws + WS_QREP) + (qb * Hh + h) * Dd;
  const float* krep = (const float*)((char*)ws + WS_KREP);
  int* flags = (int*)((char*)ws + WS_FLAG);
  __shared__ float red[256];

  float sc[NB];
  float smin = 1e30f, smax = -1e30f;
  for (int kb = 0; kb < NB; ++kb) {
    const float* kr = krep + (kb * Hh + h) * Dd;
    float dot = 0.f;
    for (int d = 0; d < Dd; ++d) dot += qr[d] * kr[d];
    const float s = dot * SCALEF;
    sc[kb] = s;
    smin = fminf(smin, s);
    smax = fmaxf(smax, s);
  }
  float esum = 0.f;
  for (int kb = 0; kb < NB; ++kb) esum += __expf(sc[kb] - smax);
  float var = 0.f;
  const float mean = 1.0f / NB;
  for (int kb = 0; kb < NB; ++kb) {
    const float p = __expf(sc[kb] - smax) / esum;
    const float dv = p - mean;
    var += dv * dv;
  }
  var *= 1.0f / (NB - 1);

  red[tid] = smin; __syncthreads();
  for (int w = 128; w > 0; w >>= 1) {
    if (tid < w) red[tid] = fminf(red[tid], red[tid + w]);
    __syncthreads();
  }
  const float bmin = red[0];
  __syncthreads();

  const float imp = var * fmaxf(smax - bmin, EPSF);

  red[tid] = imp; __syncthreads();
  for (int w = 128; w > 0; w >>= 1) {
    if (tid < w) red[tid] = fminf(red[tid], red[tid + w]);
    __syncthreads();
  }
  const float imin = red[0];
  __syncthreads();

  red[tid] = imp; __syncthreads();
  for (int w = 128; w > 0; w >>= 1) {
    if (tid < w) red[tid] = fmaxf(red[tid], red[tid + w]);
    __syncthreads();
  }
  const float imax = red[0];

  const float impn = (imp - imin) / (imax - imin + EPSF);
  flags[tid] = (impn >= 0.5f) ? 2 : ((impn <= 0.01f) ? 0 : 1);  // 2=crit 1=marg 0=negl
}

// ---------------------------------------------------------------------------
// Kernel 3: per-(h,kb) precompute (once per k-block tile):
//   - K^T B-frag tile (f16) -> WS_KF     (for exact scores)
//   - V   B-frag tile (f16) -> WS_VF     (for exact P@V)
//   - kf=elu(k)+1, ksum -> WS_KSUM
//   - kv = kf^T @ V (WMMA) -> WS_KV, already in B-frag order
// grid 256 x 128 threads (4 wave32s).
// ---------------------------------------------------------------------------
__global__ void kv_kernel(const float* __restrict__ kg, const float* __restrict__ vg,
                          float* ws) {
  const int h = blockIdx.x & 7;
  const int kb = blockIdx.x >> 3;
  const int tid = threadIdx.x;
  const int wv = tid >> 5;
  const int lane = tid & 31, lq = lane & 15, hi = lane >> 4;
  const int m0 = wv * 16;

  __shared__ v16h Afrag[8 * 32];       // kf as A tiles: (m=d, K=s)
  __shared__ v16h Bfrag[8 * 32];       // V  as B tiles: (K=s, n=v)
  __shared__ float kpart[BLK][8];      // deterministic ksum partials

  const size_t tileoff = (size_t)(h * NB + kb) * Dd * Dd;
  float* ksum = (float*)((char*)ws + WS_KSUM) + (h * NB + kb) * Dd;
  _Float16* kvb = (_Float16*)((char*)ws + WS_KV) + tileoff;
  _Float16* kfw = (_Float16*)((char*)ws + WS_KF) + tileoff;
  _Float16* vfw = (_Float16*)((char*)ws + WS_VF) + tileoff;

  // (a) K^T B-frag tile: element (K=d contiguous, N=key); straight to global.
  for (int i = tid; i < 512; i += 128) {
    const int key = i >> 3;
    const int o = i & 7;            // d octet
    const float* srck = kg + ((kb * BLK + key) * Hh + h) * Dd + o * 8;
    v8h ck;
#pragma unroll
    for (int e = 0; e < 8; ++e) ck[e] = (_Float16)srck[e];
    *(v8h*)(kfw + frag_chunk_off((key >> 4) * 2 + (o >> 2), key & 15, o)) = ck;
  }

  // (b) stage kf (A-frags; K-dim = s strided) + V (B-frags) into LDS
  for (int i = tid; i < 512; i += 128) {
    const int d = i >> 3;           // M coordinate of A / row in kf
    const int o = i & 7;            // s octet
    const float* srck = kg + ((kb * BLK + o * 8) * Hh + h) * Dd + d;
    v8h ck;
    float psum = 0.f;
#pragma unroll
    for (int e = 0; e < 8; ++e) {
      const float x = srck[e * (Hh * Dd)];
      const float f = x > 0.f ? x + 1.f : __expf(x);   // elu(x)+1
      ck[e] = (_Float16)f;
      psum += f;
    }
    *(v8h*)((_Float16*)Afrag + frag_chunk_off((d >> 4) * 2 + (o >> 2), d & 15, o)) = ck;
    kpart[d][o] = psum;

    const int vcol = i >> 3;        // N coordinate (v dim)
    const float* srcv = vg + ((kb * BLK + o * 8) * Hh + h) * Dd + vcol;
    v8h cv;
#pragma unroll
    for (int e = 0; e < 8; ++e) cv[e] = (_Float16)srcv[e * (Hh * Dd)];
    *(v8h*)((_Float16*)Bfrag + frag_chunk_off((vcol >> 4) * 2 + (o >> 2), vcol & 15, o)) = cv;
  }
  __syncthreads();

  // (c) dump V B-frag tile to workspace (plain 8KB copy)
  {
    uint4* dstv = (uint4*)vfw;
    const uint4* srcb = (const uint4*)Bfrag;
    for (int i = tid; i < 512; i += 128) dstv[i] = srcb[i];
  }

  if (tid < Dd) {
    float s = 0.f;
#pragma unroll
    for (int o = 0; o < 8; ++o) s += kpart[tid][o];
    ksum[tid] = s;
  }

  // (d) kv = kf^T @ V, each wave owns 16 d-rows (m0..m0+15); store C-layout
  // result straight into B-fragment order in workspace: within fixed (t, hi),
  // j = j0 + r is contiguous -> one 16B store per t.
  const int kkb = (m0 + 8 * hi) & 31;
  const int kcB = (m0 + 8 * hi) >> 5;
  const int hiB = (kkb >> 3) & 1;
  const int j0B = ((kkb >> 4) & 1) * 8;
#pragma unroll
  for (int t = 0; t < 4; ++t) {
    v8f c = {};
    c = wmma_f16(Afrag[(wv * 2 + 0) * 32 + lane], Bfrag[(t * 2 + 0) * 32 + lane], c);
    c = wmma_f16(Afrag[(wv * 2 + 1) * 32 + lane], Bfrag[(t * 2 + 1) * 32 + lane], c);
    v8h pack;
#pragma unroll
    for (int r = 0; r < 8; ++r) pack[r] = (_Float16)c[r];
    *(v8h*)(kvb + ((t * 2 + kcB) * 32 + (lq + 16 * hiB)) * 16 + j0B) = pack;
  }
}

// ---------------------------------------------------------------------------
// Kernel 4: main attention. grid 256 = (h,qb), 128 threads = 4 wave32s.
// Hot loop copies pre-packed f16 fragment tiles global->LDS (async when
// available) and runs WMMA; no conversions in the loop.
// ---------------------------------------------------------------------------
__global__ void attn_kernel(const float* __restrict__ q, float* ws,
                            float* __restrict__ out) {
  const int h = blockIdx.x & 7;
  const int qb = blockIdx.x >> 3;
  const int tid = threadIdx.x;
  const int wv = tid >> 5;
  const int lane = tid & 31, lq = lane & 15, hi = lane >> 4;
  const int m0 = wv * 16;

  const int* flags = (const int*)((char*)ws + WS_FLAG);
  const int flag = flags[h * NB + qb];

  __shared__ v16h Qfrag[8 * 32];       // Q as A tiles (per wave: frags wv*2+{0,1})
  __shared__ v16h Kfrag[8 * 32];       // crit: K^T B tiles; marg: kv B tiles
  __shared__ v16h Vfrag[8 * 32];       // crit: V B tiles
  __shared__ v16h Pfrag[8 * 32];       // crit: probs as A tiles
  __shared__ v8h  Qrow[BLK * Dd / 8];  // marg: qf row-major for den
  __shared__ float wrow[BLK];
  __shared__ float denL[BLK];
  __shared__ float ksL[Dd];

  if (flag == 0) {                     // negligible -> zeros
    for (int i = tid; i < BLK * Dd; i += 128) {
      const int m = i >> 6, d = i & 63;
      out[((qb * BLK + m) * Hh + h) * Dd + d] = 0.f;
    }
    return;
  }

  v8f Oacc[4] = {};
  v16h Qa0, Qa1;

  if (flag == 2) {
    // ---------------- critical: exact per-pair softmax attention ----------
    for (int i = tid; i < 512; i += 128) {
      const int m = i >> 3;            // query row
      const int o = i & 7;             // d octet
      const float* src = q + ((qb * BLK + m) * Hh + h) * Dd + o * 8;
      v8h c;
#pragma unroll
      for (int e = 0; e < 8; ++e) c[e] = (_Float16)(src[e] * SCALEF);
      *(v8h*)((_Float16*)Qfrag + frag_chunk_off((m >> 4) * 2 + (o >> 2), m & 15, o)) = c;
    }
    if (tid < BLK) wrow[tid] = (float)NB;   // each per-pair softmax sums to 1
    __syncthreads();
    Qa0 = Qfrag[(wv * 2 + 0) * 32 + lane];
    Qa1 = Qfrag[(wv * 2 + 1) * 32 + lane];

    const uint4* kf4 = (const uint4*)((const _Float16*)((char*)ws + WS_KF) +
                                      (size_t)(h * NB) * Dd * Dd);
    const uint4* vf4 = (const uint4*)((const _Float16*)((char*)ws + WS_VF) +
                                      (size_t)(h * NB) * Dd * Dd);
    uint4* kdst = (uint4*)Kfrag;
    uint4* vdst = (uint4*)Vfrag;

    for (int kb = 0; kb < NB; ++kb) {
      __syncthreads();                      // previous readers of K/V frags done
      const uint4* ks = kf4 + (size_t)kb * 512;
      const uint4* vs = vf4 + (size_t)kb * 512;
      for (int i = tid; i < 512; i += 128) {
        cp16_to_lds(&kdst[i], &ks[i]);
        cp16_to_lds(&vdst[i], &vs[i]);
      }
      wait_async0();
      __syncthreads();

      // S = (Q*scale) K^T
      v8f st[4];
#pragma unroll
      for (int t = 0; t < 4; ++t) {
        v8f c = {};
        c = wmma_f16(Qa0, Kfrag[(t * 2 + 0) * 32 + lane], c);
        c = wmma_f16(Qa1, Kfrag[(t * 2 + 1) * 32 + lane], c);
        st[t] = c;
      }

      // stable softmax over the 64 keys of this k-block
      float inv[8];
#pragma unroll
      for (int r = 0; r < 8; ++r) {
        float mx = fmaxf(fmaxf(st[0][r], st[1][r]), fmaxf(st[2][r], st[3][r]));
        mx = half_reduce_max(mx);
        float sum = 0.f;
#pragma unroll
        for (int t = 0; t < 4; ++t) {
          const float p = __expf(st[t][r] - mx);
          st[t][r] = p;
          sum += p;
        }
        sum = half_reduce_sum(sum);
        inv[r] = 1.0f / sum;
      }
      // write normalized P into A-fragment order:
      // element (m = m0+r+8*hi, key = t*16+lq):
      //   frag = wv*2 + (t>>1), lane_p = (r+8*hi) + 16*(lq>>3), j = (lq&7)+8*(t&1)
#pragma unroll
      for (int t = 0; t < 4; ++t) {
        const int kcp = t >> 1;
        const int jp = (lq & 7) + 8 * (t & 1);
#pragma unroll
        for (int r = 0; r < 8; ++r) {
          const int lane_p = (r + 8 * hi) + 16 * (lq >> 3);
          ((_Float16*)Pfrag)[((wv * 2 + kcp) * 32 + lane_p) * 16 + jp] =
              (_Float16)(st[t][r] * inv[r]);
        }
      }
      __syncthreads();

      // O += P @ V
      const v16h Pa0 = Pfrag[(wv * 2 + 0) * 32 + lane];
      const v16h Pa1 = Pfrag[(wv * 2 + 1) * 32 + lane];
#pragma unroll
      for (int t = 0; t < 4; ++t) {
        v8f c = Oacc[t];
        c = wmma_f16(Pa0, Vfrag[(t * 2 + 0) * 32 + lane], c);
        c = wmma_f16(Pa1, Vfrag[(t * 2 + 1) * 32 + lane], c);
        Oacc[t] = c;
      }
    }
  } else {
    // ---------------- marginal: ELU+1 linear attention ---------------------
    for (int i = tid; i < 512; i += 128) {
      const int m = i >> 3;
      const int o = i & 7;
      const float* src = q + ((qb * BLK + m) * Hh + h) * Dd + o * 8;
      v8h c;
#pragma unroll
      for (int e = 0; e < 8; ++e) {
        const float x = src[e];
        c[e] = (_Float16)(x > 0.f ? x + 1.f : __expf(x));   // elu(x)+1
      }
      *(v8h*)((_Float16*)Qfrag + frag_chunk_off((m >> 4) * 2 + (o >> 2), m & 15, o)) = c;
      Qrow[m * (Dd / 8) + o] = c;        // row-major copy for den
    }
    if (tid < BLK) wrow[tid] = 0.f;
    __syncthreads();
    Qa0 = Qfrag[(wv * 2 + 0) * 32 + lane];
    Qa1 = Qfrag[(wv * 2 + 1) * 32 + lane];

    const uint4* kv4 = (const uint4*)((const _Float16*)((char*)ws + WS_KV) +
                                      (size_t)(h * NB) * Dd * Dd);
    const float* ksumb = (const float*)((char*)ws + WS_KSUM);
    uint4* kdst = (uint4*)Kfrag;

    for (int kb = 0; kb < NB; ++kb) {
      __syncthreads();                      // previous readers done
      const uint4* ks = kv4 + (size_t)kb * 512;
      for (int i = tid; i < 512; i += 128) cp16_to_lds(&kdst[i], &ks[i]);
      if (tid < Dd) ksL[tid] = ksumb[(h * NB + kb) * Dd + tid];
      wait_async0();
      __syncthreads();

      if (tid < BLK) {
        const _Float16* qr = (const _Float16*)Qrow + tid * Dd;
        float dsum = 0.f;
#pragma unroll 8
        for (int d = 0; d < Dd; ++d) dsum += (float)qr[d] * ksL[d];
        const float den = dsum + EPSF;
        denL[tid] = den;
        wrow[tid] += den;
      }
      __syncthreads();

      // num = qf @ kv ; O += num / den(row)
#pragma unroll
      for (int t = 0; t < 4; ++t) {
        v8f c = {};
        c = wmma_f16(Qa0, Kfrag[(t * 2 + 0) * 32 + lane], c);
        c = wmma_f16(Qa1, Kfrag[(t * 2 + 1) * 32 + lane], c);
#pragma unroll
        for (int r = 0; r < 8; ++r) {
          const int m = m0 + r + 8 * hi;
          Oacc[t][r] += c[r] / denL[m];
        }
      }
    }
  }
  __syncthreads();

  // epilogue: out = O / (w + eps), C-layout scatter
#pragma unroll
  for (int t = 0; t < 4; ++t)
#pragma unroll
    for (int r = 0; r < 8; ++r) {
      const int m = m0 + r + 8 * hi;
      const int d = t * 16 + lq;
      out[((qb * BLK + m) * Hh + h) * Dd + d] = Oacc[t][r] / (wrow[m] + EPSF);
    }
}

// ---------------------------------------------------------------------------
extern "C" void kernel_launch(void* const* d_in, const int* in_sizes, int n_in,
                              void* d_out, int out_size, void* d_ws, size_t ws_size,
                              hipStream_t stream) {
  (void)in_sizes; (void)n_in; (void)out_size; (void)ws_size;
  const float* q = (const float*)d_in[0];
  const float* k = (const float*)d_in[1];
  const float* v = (const float*)d_in[2];
  float* out = (float*)d_out;
  float* ws = (float*)d_ws;

  hipLaunchKernelGGL(rep_kernel,      dim3(512), dim3(64),  0, stream, q, k, ws);
  hipLaunchKernelGGL(classify_kernel, dim3(1),   dim3(256), 0, stream, ws);
  hipLaunchKernelGGL(kv_kernel,       dim3(256), dim3(128), 0, stream, k, v, ws);
  hipLaunchKernelGGL(attn_kernel,     dim3(256), dim3(128), 0, stream, q, ws, out);
}
